// HungarianMatcher_51118700757221
// MI455X (gfx1250) — matrix operand order repaired and checked by reading, not compiled, not run on hardware
//
#include <hip/hip_runtime.h>
#include <hip/hip_bf16.h>

// ---------------------------------------------------------------------------
// HungarianMatcher on MI455X (gfx1250)
//  Phase 1: C[b,q,t] = sum_d |out[b,q,d]-tgt[b,t,d]|  via v_wmma_f32_16x16x32_f16
//           (K = 8 targets x 4 dims; B operand is a constant 0/1 selector so the
//            matrix pipe performs the exact (t,d)->t reduction; f32 accumulate)
//  Phase 2: Jonker-Volgenant shortest-augmenting-path LSA, one workgroup per
//           batch, all solver state in LDS (320KB/WGP on CDNA5), cost rows
//           recomputed on the fly from L2-resident inputs.
//  NOTE: |x| must be emitted via inline asm: both llvm.fabs and the
//        maxnum(x,-x) canonicalization hit a packed-f32 fabs ISel crash in
//        this gfx1250 toolchain. v_max_num_f32 with a neg source modifier is
//        opaque to InstCombine/SLP and lowers cleanly.
// ---------------------------------------------------------------------------

typedef __attribute__((ext_vector_type(16))) _Float16 v16h;
typedef __attribute__((ext_vector_type(8)))  float    v8f;
typedef __attribute__((ext_vector_type(4)))  float    v4f;

#define NB   16
#define NQ   4096
#define NTGT 128
#define ND   4

static __device__ __forceinline__ float absdiff(float a, float b) {
  float d = a - b;
  float r;
  // |d| = max(d, -d); '-' modifier forces VOP3 encoding, opaque to the optimizer
  asm("v_max_num_f32 %0, %1, -%1" : "=v"(r) : "v"(d));
  return r;
}

// ============================ Phase 1: cost matrix ==========================
// Grid: (NQ/128, NB) blocks of 256 threads (8 waves). Each wave owns a 16-row
// q-tile and loops over the 8 t-tiles; 2 WMMAs per 16x16 C tile.
__global__ __launch_bounds__(256)
void hm_cost_wmma(const float* __restrict__ outputs,
                  const float* __restrict__ targets,
                  float* __restrict__ C) {
  __shared__ float tg[NTGT * ND];                 // 2KB target stage
  const int b    = blockIdx.y;
  const int qblk = blockIdx.x;                    // 128 q per block
  const int tid  = threadIdx.x;

  for (int k = tid; k < NTGT * ND; k += 256)
    tg[k] = targets[(size_t)b * NTGT * ND + k];
  __syncthreads();

  const int wave = tid >> 5;
  const int lane = tid & 31;
  const int g    = lane >> 4;                     // lane half (A/B K-split)
  const int n    = lane & 15;                     // A row (M) and B column (N)
  const int q    = qblk * 128 + wave * 16 + n;

  // per-lane outputs row (16B coalesced)
  v4f o = *(const v4f*)(outputs + ((size_t)b * NQ + q) * ND);
  float od[4] = {o.x, o.y, o.z, o.w};

  // Constant selector B matrices: B[(t,d), n] = (t == n) for the two t-halves.
  // B 32x16 f16 layout: lane half g holds K = (e<8) ? 8g+e : 16+8g+(e-8) (2 per VGPR, lo first).
  v16h b1, b2;
#pragma unroll
  for (int e = 0; e < 16; ++e) {
    int k  = (e < 8) ? (8 * g + e) : (16 + 8 * g + (e - 8));
    int tl = k >> 2;                              // t-local = K/4
    b1[e] = (_Float16)((n <  8 && tl == n      ) ? 1.0f : 0.0f);
    b2[e] = (_Float16)((n >= 8 && tl == (n - 8)) ? 1.0f : 0.0f);
  }

  float* Cb = C + (size_t)b * NQ * NTGT;

#pragma unroll 1
  for (int tt = 0; tt < 8; ++tt) {
    const int tbase = tt * 16;
    // A 16x32 f16 layout: lane half g holds K = (e<8) ? 8g+e : 16+8g+(e-8)
    v16h a1, a2;
#pragma unroll
    for (int e = 0; e < 16; ++e) {
      int K  = (e < 8) ? (8 * g + e) : (16 + 8 * g + (e - 8));
      int tl = K >> 2;
      int d  = K & 3;
      a1[e] = (_Float16)absdiff(od[d], tg[(tbase     + tl) * ND + d]);
      a2[e] = (_Float16)absdiff(od[d], tg[(tbase + 8 + tl) * ND + d]);
    }
    v8f c = {};
    c = __builtin_amdgcn_wmma_f32_16x16x32_f16(false, a1, false, b1,
                                               (short)0, c, false, false);
    c = __builtin_amdgcn_wmma_f32_16x16x32_f16(false, a2, false, b2,
                                               (short)0, c, false, false);
    // D layout: VGPR r, lanes0-15 -> M=r, lanes16-31 -> M=8+r; N = lane%16
#pragma unroll
    for (int r = 0; r < 8; ++r) {
      int qq = qblk * 128 + wave * 16 + r + 8 * g;
      Cb[(size_t)qq * NTGT + tbase + n] = c[r];
    }
  }
}

// ============================ Phase 2: JV LSA ===============================
#define NTH 256
__global__ __launch_bounds__(NTH)
void hm_lsa(const float* __restrict__ outputs,
            const float* __restrict__ targets,
            float* __restrict__ out_base) {
  const int b   = blockIdx.x;
  const int tid = threadIdx.x;

  __shared__ float          tg[NTGT * ND];        //  2 KB
  __shared__ float          shortestA[NQ];        // 16 KB
  __shared__ float          vA[NQ];               // 16 KB
  __shared__ short          predA[NQ];            //  8 KB
  __shared__ short          row4colA[NQ];         //  8 KB
  __shared__ unsigned char  SCc[NQ];              //  4 KB
  __shared__ float          uA[NTGT];
  __shared__ unsigned char  SRc[NTGT];
  __shared__ short          col4rowA[NTGT];
  __shared__ float          redV[NTH];
  __shared__ int            redI[NTH];
  __shared__ int            sh_i, sh_sink;
  __shared__ float          sh_min;

  const float* outb = outputs + (size_t)b * NQ * ND;
  __builtin_prefetch(outb + (size_t)tid * 64, 0, 1);   // global_prefetch_b8

  for (int k = tid; k < NTGT * ND; k += NTH)
    tg[k] = targets[(size_t)b * NTGT * ND + k];
  for (int j = tid; j < NQ; j += NTH) { vA[j] = 0.f; row4colA[j] = -1; }
  for (int i = tid; i < NTGT; i += NTH) { uA[i] = 0.f; col4rowA[i] = -1; }
  __syncthreads();

  for (int cur_row = 0; cur_row < NTGT; ++cur_row) {
    for (int j = tid; j < NQ; j += NTH) { shortestA[j] = __builtin_inff(); SCc[j] = 0; }
    for (int i = tid; i < NTGT; i += NTH) SRc[i] = 0;
    if (tid == 0) { sh_i = cur_row; sh_sink = -1; sh_min = 0.f; }
    __syncthreads();

    for (;;) {
      const int   i    = sh_i;
      if (tid == 0) SRc[i] = 1;
      const float ui   = uA[i];
      const float minv = sh_min;
      const float t0 = tg[i*ND+0], t1 = tg[i*ND+1], t2 = tg[i*ND+2], t3 = tg[i*ND+3];

      float bestv = __builtin_inff();
      int   bestj = NQ;
      for (int j = tid; j < NQ; j += NTH) {
        if (!SCc[j]) {
          float s = shortestA[j];
          const float* op = outb + (size_t)j * ND;
          float cst = absdiff(op[0], t0) + absdiff(op[1], t1);
          cst = cst + absdiff(op[2], t2) + absdiff(op[3], t3);
          float cand = minv + cst - ui - vA[j];
          if (cand < s) { s = cand; shortestA[j] = cand; predA[j] = (short)i; }
          if (s < bestv || (s == bestv && j < bestj)) { bestv = s; bestj = j; }
        }
      }
      redV[tid] = bestv; redI[tid] = bestj;
      __syncthreads();
      for (int off = NTH / 2; off > 0; off >>= 1) {
        if (tid < off) {
          float ov = redV[tid + off]; int oj = redI[tid + off];
          if (ov < redV[tid] || (ov == redV[tid] && oj < redI[tid])) {
            redV[tid] = ov; redI[tid] = oj;
          }
        }
        __syncthreads();
      }
      if (tid == 0) {
        const int jm = redI[0];
        sh_min = redV[0];
        SCc[jm] = 1;
        const short r = row4colA[jm];
        if (r < 0) sh_sink = jm; else sh_i = r;
      }
      __syncthreads();
      if (sh_sink >= 0) break;
    }

    // dual updates
    const float minv = sh_min;
    for (int ip = tid; ip < NTGT; ip += NTH) {
      if (ip == cur_row)      uA[ip] += minv;
      else if (SRc[ip])       uA[ip] += minv - shortestA[col4rowA[ip]];
    }
    for (int j = tid; j < NQ; j += NTH)
      if (SCc[j]) vA[j] -= minv - shortestA[j];
    __syncthreads();

    // augment alternating path (tiny pointer chase; thread 0)
    if (tid == 0) {
      int j = sh_sink;
      for (;;) {
        const int i = predA[j];
        row4colA[j] = (short)i;
        const int jn = col4rowA[i];
        col4rowA[i] = (short)j;
        j = jn;
        if (i == cur_row) break;
      }
    }
    __syncthreads();
  }

  // transposed-output fixup: order = argsort(col4row); row_ind = col4row[order],
  // col_ind = order. Values are distinct -> rank = count(smaller).
  long long* rowi = (long long*)(out_base + (size_t)NB * NQ * NTGT);
  long long* coli = rowi + (size_t)NB * NTGT;
  if (tid < NTGT) {
    const int key = col4rowA[tid];
    int rank = 0;
    for (int s = 0; s < NTGT; ++s) rank += (col4rowA[s] < key);
    rowi[(size_t)b * NTGT + rank] = (long long)key;
    coli[(size_t)b * NTGT + rank] = (long long)tid;
  }
}

// ================================ launcher ==================================
extern "C" void kernel_launch(void* const* d_in, const int* in_sizes, int n_in,
                              void* d_out, int out_size, void* d_ws, size_t ws_size,
                              hipStream_t stream) {
  const float* outputs = (const float*)d_in[0];   // [16,4096,4] f32
  const float* targets = (const float*)d_in[1];   // [16,128,4]  f32
  float* C = (float*)d_out;                       // [16,4096,128] f32 + int64 indices

  dim3 g1(NQ / 128, NB);
  hm_cost_wmma<<<g1, 256, 0, stream>>>(outputs, targets, C);
  hm_lsa<<<NB, NTH, 0, stream>>>(outputs, targets, (float*)d_out);
}
